// EdgeNet_21157008900557
// MI455X (gfx1250) — compile-verified, matrix operand order checked by reading, not run on hardware
//
#include <hip/hip_runtime.h>
#include <hip/hip_bf16.h>
#include <math.h>

typedef __attribute__((ext_vector_type(16))) _Float16 v16h;
typedef __attribute__((ext_vector_type(8)))  float    v8f;

#ifndef EPSN
#define EPSN 1e-12f
#endif

// ---------------------------------------------------------------------------
// Stage 1: fn = x / max(||x||, eps)   (one thread per node)
// ---------------------------------------------------------------------------
__global__ void fn_kernel(const float* __restrict__ x, float* __restrict__ fn, int n) {
  int i = blockIdx.x * blockDim.x + threadIdx.x;
  if (i >= n) return;
  float s = 0.f;
#pragma unroll
  for (int k = 0; k < 32; ++k) { float v = x[i * 32 + k]; s += v * v; }
  float nrm = sqrtf(s);
  float inv = 1.f / fmaxf(nrm, EPSN);
#pragma unroll
  for (int k = 0; k < 32; ++k) fn[i * 32 + k] = x[i * 32 + k] * inv;
}

// ---------------------------------------------------------------------------
// Stage 2a: neigh1[:, k<32] via reverse suffix scan (complete tril graph =>
// mean over in-edges of col c is a suffix mean over rows r > c).
// neigh1[c,k] = fn[c,k] * sum_{r>c} fn[r,k]*x[r,k] / max(n-1-c, 1)
// ---------------------------------------------------------------------------
__global__ void neigh1_scan_kernel(const float* __restrict__ x, const float* __restrict__ fn,
                                   float* __restrict__ neigh1, int n) {
  int k = threadIdx.x;  // 0..31
  float acc = 0.f;
  for (int c = n - 1; c >= 0; --c) {
    int   d   = n - 1 - c;
    float deg = (float)(d > 1 ? d : 1);
    neigh1[c * 35 + k] = fn[c * 32 + k] * acc / deg;
    acc += fn[c * 32 + k] * x[c * 32 + k];
  }
}

// ---------------------------------------------------------------------------
// Stage 2b: neigh1[:, 32+d] = sum_{r>c} |cen[c,d]-cen[r,d]| * cen[r,d] / deg
// ---------------------------------------------------------------------------
__global__ void neigh1_abs_kernel(const float* __restrict__ cen, float* __restrict__ neigh1, int n) {
  int t = blockIdx.x * blockDim.x + threadIdx.x;
  if (t >= n * 3) return;
  int c = t % n, d = t / n;
  float cc = cen[c * 3 + d];
  float s = 0.f;
  for (int r = c + 1; r < n; ++r) {
    float cr = cen[r * 3 + d];
    s += fabsf(cc - cr) * cr;
  }
  int   dd  = n - 1 - c;
  float deg = (float)(dd > 1 ? dd : 1);
  neigh1[c * 35 + 32 + d] = s / deg;
}

// ---------------------------------------------------------------------------
// Stage 3: h1 = h @ W1s + neigh1 @ W1n + b1     (h = [x | centroids], K=35)
// ---------------------------------------------------------------------------
__global__ void h1_kernel(const float* __restrict__ x, const float* __restrict__ cen,
                          const float* __restrict__ neigh1,
                          const float* __restrict__ W1s, const float* __restrict__ W1n,
                          const float* __restrict__ b1, float* __restrict__ h1, int n) {
  int t = blockIdx.x * blockDim.x + threadIdx.x;
  if (t >= n * 64) return;
  int i = t >> 6, o = t & 63;
  float s = b1[o];
#pragma unroll 5
  for (int k = 0; k < 35; ++k) {
    float hk = (k < 32) ? x[i * 32 + k] : cen[i * 3 + (k - 32)];
    s += hk * W1s[k * 64 + o] + neigh1[i * 35 + k] * W1n[k * 64 + o];
  }
  h1[t] = s;
}

// ---------------------------------------------------------------------------
// Stage 4: neigh2[c] = suffix-mean of h1 over rows r > c.
// ---------------------------------------------------------------------------
__global__ void neigh2_scan_kernel(const float* __restrict__ h1, float* __restrict__ neigh2, int n) {
  int k = threadIdx.x;  // 0..63
  float acc = 0.f;
  for (int c = n - 1; c >= 0; --c) {
    int   d   = n - 1 - c;
    float deg = (float)(d > 1 ? d : 1);
    neigh2[c * 64 + k] = acc / deg;
    acc += h1[c * 64 + k];
  }
}

// ---------------------------------------------------------------------------
// Stage 5: h2 = h1 @ W2s + neigh2 @ W2n + b2
// ---------------------------------------------------------------------------
__global__ void h2_kernel(const float* __restrict__ h1, const float* __restrict__ neigh2,
                          const float* __restrict__ W2s, const float* __restrict__ W2n,
                          const float* __restrict__ b2, float* __restrict__ h2, int n) {
  int t = blockIdx.x * blockDim.x + threadIdx.x;
  if (t >= n * 32) return;
  int i = t >> 5, o = t & 31;
  float s = b2[o];
#pragma unroll 8
  for (int k = 0; k < 64; ++k)
    s += h1[i * 64 + k] * W2s[k * 32 + o] + neigh2[i * 64 + k] * W2n[k * 32 + o];
  h2[t] = s;
}

// ---------------------------------------------------------------------------
// Stage 6: per-node factorization of the edge layer:
//   AB[i][0:32)  = A[i] = h2[i] @ W3[0:32, :]
//   AB[i][32:64) = B[i] = h2[i] @ W3[32:64, :]
// ---------------------------------------------------------------------------
__global__ void ab_kernel(const float* __restrict__ h2, const float* __restrict__ W3,
                          float* __restrict__ AB, int n) {
  int t = blockIdx.x * blockDim.x + threadIdx.x;
  if (t >= n * 64) return;
  int i = t >> 6, o = t & 63;
  int col  = o & 31;
  int base = (o < 32) ? 0 : 32;
  float s = 0.f;
#pragma unroll 8
  for (int k = 0; k < 32; ++k) s += h2[i * 32 + k] * W3[(base + k) * 32 + col];
  AB[t] = s;
}

// ---------------------------------------------------------------------------
// Stage 7 (dominant): 16-edge tiles through v_wmma_f32_16x16x32_f16.
// Integer-exact tril decode: f32 sqrt estimate (one v_sqrt_f32, TRANS pipe)
// + int32 fixup loops. Valid because E = N(N-1)/2 < 2^31 and r < 2^16, so
// r*(r-1) fits in int32; the f32 estimate is within +-2 of the true row.
// ---------------------------------------------------------------------------
__device__ __forceinline__ v8f tile_wmma(int e0, int eMax,
                                         const float* __restrict__ AB,
                                         const float* __restrict__ sb3,
                                         v16h bfrag, int m, int hi) {
  int e = e0 + m;
  e = (e > eMax) ? eMax : e;
  float ef = (float)e;
  int r = (int)((1.0f + sqrtf(fmaf(8.0f, ef, 1.0f))) * 0.5f);
  while ((r * (r - 1)) >> 1 > e) --r;
  while (((r + 1) * r) >> 1 <= e) ++r;
  int c = e - ((r * (r - 1)) >> 1);

  const float* Ac = AB + (long long)c * 64;       // A[col]
  const float* Br = AB + (long long)r * 64 + 32;  // B[row]

  v16h afrag;
#pragma unroll
  for (int j = 0; j < 16; ++j) {
    int   k  = 8 * hi + (j & 7) + 16 * (j >> 3);
    float tv = Ac[k] + Br[k] + sb3[k];
    afrag[j] = (_Float16)(tv > 0.f ? tv : 0.f);
  }
  v8f z = {};
  return __builtin_amdgcn_wmma_f32_16x16x32_f16(false, afrag, false, bfrag,
                                                (short)0, z, false, false);
}

__device__ __forceinline__ void tile_store_guarded(v8f acc, int e0, int E,
                                                   const float* __restrict__ sb4,
                                                   float* __restrict__ out, int m, int hi) {
  if (m < 2) {
    float bb = sb4[m];
#pragma unroll
    for (int v = 0; v < 8; ++v) {
      int ee = e0 + v + 8 * hi;
      if (ee < E) {
        float o = acc[v] + bb;
        out[(long long)ee * 2 + m] = (o > 0.f) ? o : 0.f;
      }
    }
  }
}

// wave32 WMMA layouts (ISA 7.12.2):
//   A frag : lane l -> M = l&15, slot j -> K = 8*(l>>4) + (j&7) + 16*(j>>3)
//   B frag : lane l -> N = l&15, same K(j) mapping
//   D (f32): VGPR v -> M = v + 8*(l>>4), N = l&15
// Two consecutive tiles per iteration -> 2 back-to-back WMMAs (ILP over the
// WMMA hazard window) and 64 contiguous output floats, stored via an LDS
// transpose as one coalesced global_store_b64 per lane (256B/wave).
__global__ void edge_mlp_kernel(const float* __restrict__ AB, const float* __restrict__ b3,
                                const float* __restrict__ W4, const float* __restrict__ b4,
                                float* __restrict__ out, int E, int T) {
  __shared__ float sb3[32];
  __shared__ float sW4[64];
  __shared__ float sb4[2];
  __shared__ float xpose[8][64];  // per-wave transpose staging (blockDim = 256 = 8 waves)
  if (threadIdx.x < 32) sb3[threadIdx.x] = b3[threadIdx.x];
  if (threadIdx.x < 64) sW4[threadIdx.x] = W4[threadIdx.x];
  if (threadIdx.x < 2)  sb4[threadIdx.x] = b4[threadIdx.x];
  __syncthreads();

  const int lane = threadIdx.x & 31;
  const int m    = lane & 15;   // A: edge-in-tile (M); B: output column (N)
  const int hi   = lane >> 4;   // K half selector
  const int eMax = E - 1;
  float*    xp   = xpose[threadIdx.x >> 5];

  // Hoist B-fragment (W4 zero-padded to 32x16) out of the tile loop.
  v16h bfrag;
#pragma unroll
  for (int j = 0; j < 16; ++j) {
    int k = 8 * hi + (j & 7) + 16 * (j >> 3);
    bfrag[j] = (m < 2) ? (_Float16)sW4[k * 2 + m] : (_Float16)0.f;
  }

  int waveId     = blockIdx.x * (blockDim.x >> 5) + (threadIdx.x >> 5);
  int totalWaves = gridDim.x * (blockDim.x >> 5);

  for (int t = waveId * 2; t < T; t += totalWaves * 2) {
    int  e0       = t * 16;
    bool havePair = (t + 1) < T;               // wave-uniform
    bool fullPair = havePair && (e0 + 32 <= E);

    v8f acc0 = tile_wmma(e0, eMax, AB, sb3, bfrag, m, hi);

    if (fullPair) {
      v8f acc1 = tile_wmma(e0 + 16, eMax, AB, sb3, bfrag, m, hi);
      // Fast path: LDS transpose -> one coalesced float2 store per lane.
      if (m < 2) {
        float bb = sb4[m];
#pragma unroll
        for (int v = 0; v < 8; ++v) {
          int   p  = (v + 8 * hi) * 2 + m;  // position within 32-float tile
          float o0 = acc0[v] + bb;
          float o1 = acc1[v] + bb;
          xp[p]      = (o0 > 0.f) ? o0 : 0.f;
          xp[32 + p] = (o1 > 0.f) ? o1 : 0.f;
        }
      }
      asm volatile("s_wait_dscnt 0x0" ::: "memory");
      float2 pr = ((const float2*)xp)[lane];
      *(float2*)(out + (long long)e0 * 2 + lane * 2) = pr;
      asm volatile("" ::: "memory");  // keep next iter's DS stores after this load
    } else {
      // Cold tail: guarded scalar stores.
      tile_store_guarded(acc0, e0, E, sb4, out, m, hi);
      if (havePair) {
        v8f acc1 = tile_wmma(e0 + 16, eMax, AB, sb3, bfrag, m, hi);
        tile_store_guarded(acc1, e0 + 16, E, sb4, out, m, hi);
      }
    }
  }
}

// ---------------------------------------------------------------------------
// Host-side launcher. Inputs (setup_inputs order, all f32):
//  0:x[N,32] 1:centroids[N,3] 2:W1s[35,64] 3:W1n[35,64] 4:b1[64]
//  5:W2s[64,32] 6:W2n[64,32] 7:b2[32] 8:W3[64,32] 9:b3[32] 10:W4[32,2] 11:b4[2]
// ---------------------------------------------------------------------------
extern "C" void kernel_launch(void* const* d_in, const int* in_sizes, int n_in,
                              void* d_out, int out_size, void* d_ws, size_t ws_size,
                              hipStream_t stream) {
  const float* x   = (const float*)d_in[0];
  const float* cen = (const float*)d_in[1];
  const float* W1s = (const float*)d_in[2];
  const float* W1n = (const float*)d_in[3];
  const float* b1  = (const float*)d_in[4];
  const float* W2s = (const float*)d_in[5];
  const float* W2n = (const float*)d_in[6];
  const float* b2  = (const float*)d_in[7];
  const float* W3  = (const float*)d_in[8];
  const float* b3  = (const float*)d_in[9];
  const float* W4  = (const float*)d_in[10];
  const float* b4  = (const float*)d_in[11];
  float* out = (float*)d_out;

  const int n = in_sizes[0] / 32;               // 2048
  const int E = n * (n - 1) / 2;                // 2,096,128  (< 2^31)
  const int T = (E + 15) / 16;                  // 16-edge WMMA tiles

  // Workspace carve-up (floats)
  float* ws     = (float*)d_ws;
  float* fn     = ws;                          // n*32
  float* neigh1 = fn + (size_t)n * 32;         // n*35
  float* h1     = neigh1 + (size_t)n * 35;     // n*64
  float* neigh2 = h1 + (size_t)n * 64;         // n*64
  float* h2     = neigh2 + (size_t)n * 64;     // n*32
  float* ABt    = h2 + (size_t)n * 32;         // n*64

  const int B = 256;
  fn_kernel<<<(n + B - 1) / B, B, 0, stream>>>(x, fn, n);
  neigh1_scan_kernel<<<1, 32, 0, stream>>>(x, fn, neigh1, n);
  neigh1_abs_kernel<<<(3 * n + B - 1) / B, B, 0, stream>>>(cen, neigh1, n);
  h1_kernel<<<(64 * n + B - 1) / B, B, 0, stream>>>(x, cen, neigh1, W1s, W1n, b1, h1, n);
  neigh2_scan_kernel<<<1, 64, 0, stream>>>(h1, neigh2, n);
  h2_kernel<<<(32 * n + B - 1) / B, B, 0, stream>>>(h1, neigh2, W2s, W2n, b2, h2, n);
  ab_kernel<<<(64 * n + B - 1) / B, B, 0, stream>>>(h2, W3, ABt, n);

  // Dominant edge kernel: 1024 blocks x 8 waves, 2 tiles per wave-iteration.
  edge_mlp_kernel<<<1024, 256, 0, stream>>>(ABt, b3, W4, b4, out, E, T);
}
